// LabelSimilarity_1417339207722
// MI455X (gfx1250) — matrix operand order repaired and checked by reading, not compiled, not run on hardware
//
#include <hip/hip_runtime.h>
#include <math.h>

// ---------------- problem constants ----------------
#define S_TOT 256
#define B_TOT 128
#define D_DIM 768
#define L_TOT 200
#define N_TOT 1600            // L*V
#define EPS_N 1e-20f

// ---------------- shared tiling ----------------
#define BM 32                 // b rows per block
#define BN 128                // n cols per block
#define NKS 24                // 768 / 32
#define SCHUNKS 8
#define S_PER (S_TOT / SCHUNKS)    // 32
#define NT ((N_TOT + BN - 1) / BN) // 13
#define BT (B_TOT / BM)            // 4

// TDM-padded LDS row geometry: 768 bf16 = 1536 B; TDM inserts 16 B after every
// 512 B stored (pad_interval=6 -> 128 DWORDs, pad_amount=3 -> 4 DWORDs).
// Row stride = 1536 + 3*16 = 1584 B (396 DWORDs == 12 mod 64 banks).
#define ROWB 1584
// element k (bf16) byte offset within a row: 2k + 16*(k>>8)   (16B-aligned for k%8==0)

#define PART_FLOATS (SCHUNKS * B_TOT * L_TOT)   // 204800
#define NE_CNT (S_TOT * B_TOT)                  // 32768
#define EMB_ELEMS ((size_t)S_TOT * B_TOT * D_DIM)   // 25,165,824
#define LAB_ELEMS ((size_t)N_TOT * D_DIM)           // 1,228,800

typedef __bf16 bf16;
typedef bf16  v16bf  __attribute__((ext_vector_type(16)));
typedef bf16  bf16x8 __attribute__((ext_vector_type(8)));
typedef bf16  bf16x4 __attribute__((ext_vector_type(4)));
typedef float v8f    __attribute__((ext_vector_type(8)));
typedef unsigned int u32x4 __attribute__((ext_vector_type(4)));
typedef int   i32x8 __attribute__((ext_vector_type(8)));
typedef int   i32x4 __attribute__((ext_vector_type(4)));

union FragU { v16bf v; bf16x8 h[2]; };

// ---------------- row L2 norms: one wave per 768-elem row ----------------
__global__ void norms_kernel(const float* __restrict__ src, float* __restrict__ dst, int rows) {
  int w    = threadIdx.x >> 5;
  int lane = threadIdx.x & 31;
  int row  = blockIdx.x * 8 + w;
  if (row >= rows) return;
  const float* p = src + (size_t)row * D_DIM;
  float s = 0.f;
  for (int k = lane; k < D_DIM; k += 32) { float v = p[k]; s += v * v; }
  #pragma unroll
  for (int t = 16; t >= 1; t >>= 1) s += __shfl_xor(s, t, 32);
  if (lane == 0) dst[row] = sqrtf(s);
}

// ---------------- f32 -> bf16 bulk convert (8 elems / thread) ----------------
__global__ void cvt_bf16_kernel(const float* __restrict__ src, bf16* __restrict__ dst, int n8) {
  int i = blockIdx.x * blockDim.x + threadIdx.x;
  if (i >= n8) return;
  const float4* s4 = (const float4*)src;
  float4 a = s4[i * 2], b = s4[i * 2 + 1];
  bf16x8 o;
  o[0] = (bf16)a.x; o[1] = (bf16)a.y; o[2] = (bf16)a.z; o[3] = (bf16)a.w;
  o[4] = (bf16)b.x; o[5] = (bf16)b.y; o[6] = (bf16)b.z; o[7] = (bf16)b.w;
  ((bf16x8*)dst)[i] = o;
}

// ---------------- TDM: 2-D bf16 tile (rows x 768) global -> padded LDS ----------------
__device__ __forceinline__ void tdm_load_2d(unsigned int lds_off, const bf16* gptr,
                                            unsigned int rows_tensor, unsigned int rows_tile) {
  unsigned long long ga = (unsigned long long)(uintptr_t)gptr;
  u32x4 g0;
  g0[0] = 1u;                                   // count=1, user descriptor
  g0[1] = lds_off;                              // LDS byte address of tile
  g0[2] = (unsigned int)ga;                     // global_addr lo
  g0[3] = (unsigned int)(ga >> 32) | (2u << 30);// global_addr hi | type=2 (image)
  i32x8 g1;
  g1[0] = (int)((1u << 16)       // data_size = 2 bytes
              | (1u << 20)       // pad_enable
              | (6u << 22)       // pad_interval: 128 DWORDs
              | (3u << 25));     // pad_amount: 4 DWORDs (16 B, keeps b128 alignment)
  g1[1] = (int)(768u << 16);                       // tensor_dim0[15:0] = 768
  g1[2] = (int)((768u >> 16) | (rows_tensor << 16));// dim0 hi | tensor_dim1 lo
  g1[3] = (int)((rows_tensor >> 16) | (768u << 16));// dim1 hi | tile_dim0 = 768
  g1[4] = (int)(rows_tile & 0xFFFFu);              // tile_dim1 | tile_dim2=0 (2-D)
  g1[5] = 768;                                     // tensor_dim0_stride lo32
  g1[6] = 0;                                       // stride hi | dim1_stride lo
  g1[7] = 0;
  i32x4 g2; g2[0] = 1; g2[1] = 1; g2[2] = 0; g2[3] = 0;  // tensor_dim2=1, dims 2+ unused
  i32x4 g3; g3[0] = 0; g3[1] = 0; g3[2] = 0; g3[3] = 0;
  i32x8 gx; gx[0]=0; gx[1]=0; gx[2]=0; gx[3]=0; gx[4]=0; gx[5]=0; gx[6]=0; gx[7]=0; // aux (6-arg form), zero
  __builtin_amdgcn_tensor_load_to_lds(g0, g1, g2, g3, gx, 0);
}

// ================= fast path: pre-converted bf16 + TDM staging =================
__global__ __launch_bounds__(128, 1)
void sim_kernel_tdm(const bf16* __restrict__ embed_bf, const bf16* __restrict__ labels_bf,
                    const float* __restrict__ ne, const float* __restrict__ nl,
                    float* __restrict__ partial) {
  __shared__ alignas(16) unsigned char sB[BN * ROWB];       // 202,752 B (once)
  __shared__ alignas(16) unsigned char sA[2][BM * ROWB];    // 101,376 B (double-buffered)
  __shared__ alignas(16) float sInvNe[BM];

  const int tid   = threadIdx.x;
  const int lane  = tid & 31;
  const int wave  = tid >> 5;          // 0..3 -> 32-wide n quadrant
  const int hi    = (lane >> 4) & 1;
  const int l15   = lane & 15;
  const int nbase = blockIdx.x * BN;
  const int bbase = blockIdx.y * BM;
  const int sbase = blockIdx.z * S_PER;

  const unsigned int sB_off  = (unsigned int)(uintptr_t)&sB[0];
  const unsigned int sA_off0 = (unsigned int)(uintptr_t)&sA[0][0];
  const unsigned int sA_off1 = (unsigned int)(uintptr_t)&sA[1][0];

  // B tile + first A slice via TDM; OOB label rows (n >= 1600) read back as zeros
  if (wave == 0) {
    unsigned int nrows = (unsigned int)(N_TOT - nbase); if (nrows > BN) nrows = BN;
    tdm_load_2d(sB_off, labels_bf + (size_t)nbase * D_DIM, nrows, BN);
    tdm_load_2d(sA_off0, embed_bf + ((size_t)sbase * B_TOT + bbase) * D_DIM, BM, BM);
    __builtin_amdgcn_s_wait_tensorcnt(0);
  }
  __syncthreads();

  float invnl[2];
  #pragma unroll
  for (int nf = 0; nf < 2; ++nf) {
    int n = nbase + wave * 32 + nf * 16 + l15;
    invnl[nf] = (n < N_TOT) ? __builtin_amdgcn_rcpf(fmaxf(nl[n], EPS_N)) : 0.f;
  }

  float acc[2][2][8];
  #pragma unroll
  for (int mf = 0; mf < 2; ++mf)
    #pragma unroll
    for (int nf = 0; nf < 2; ++nf)
      #pragma unroll
      for (int j = 0; j < 8; ++j) acc[mf][nf][j] = -3.402823466e38f;

  for (int si = 0; si < S_PER; ++si) {
    const int s = sbase + si;
    // prefetch next s slice into the other buffer while we compute this one
    if (si + 1 < S_PER && wave == 0) {
      unsigned int dst = ((si + 1) & 1) ? sA_off1 : sA_off0;
      tdm_load_2d(dst, embed_bf + ((size_t)(s + 1) * B_TOT + bbase) * D_DIM, BM, BM);
    }
    if (tid < BM)
      sInvNe[tid] = __builtin_amdgcn_rcpf(fmaxf(ne[(size_t)s * B_TOT + bbase + tid], EPS_N));

    const unsigned char* sAb = sA[si & 1];
    v8f cz = {};
    v8f c[2][2]; c[0][0] = cz; c[0][1] = cz; c[1][0] = cz; c[1][1] = cz;

    #pragma unroll
    for (int kb = 0; kb < NKS; ++kb) {
      const int ka  = kb * 32 + hi * 8;   // A: lane<16 K{0-7,16-23}, lane>=16 K{8-15,24-31}
      const int ka2 = ka + 16;
      const int kbb = kb * 32 + hi * 8;   // B: lane<16 K{0-7,16-23}, lane>=16 K{8-15,24-31}
      const int kb2 = kbb + 16;
      FragU a[2], b[2];
      #pragma unroll
      for (int mf = 0; mf < 2; ++mf) {
        const unsigned char* rp = sAb + (mf * 16 + l15) * ROWB;
        a[mf].h[0] = *(const bf16x8*)(rp + 2 * ka  + ((ka  >> 8) << 4));
        a[mf].h[1] = *(const bf16x8*)(rp + 2 * ka2 + ((ka2 >> 8) << 4));
      }
      #pragma unroll
      for (int nf = 0; nf < 2; ++nf) {
        const unsigned char* rp = sB + (wave * 32 + nf * 16 + l15) * ROWB;
        b[nf].h[0] = *(const bf16x8*)(rp + 2 * kbb + ((kbb >> 8) << 4));
        b[nf].h[1] = *(const bf16x8*)(rp + 2 * kb2 + ((kb2 >> 8) << 4));
      }
      #pragma unroll
      for (int mf = 0; mf < 2; ++mf)
        #pragma unroll
        for (int nf = 0; nf < 2; ++nf)
          c[mf][nf] = __builtin_amdgcn_wmma_f32_16x16x32_bf16(
              false, a[mf].v, false, b[nf].v, (short)0, c[mf][nf], false, false);
    }

    // fold dots*inv_ne into running max (inv_nl applied once at the end)
    #pragma unroll
    for (int mf = 0; mf < 2; ++mf) {
      const float4 p0 = *(const float4*)&sInvNe[mf * 16 + hi * 8];
      const float4 p1 = *(const float4*)&sInvNe[mf * 16 + hi * 8 + 4];
      float inv[8] = {p0.x, p0.y, p0.z, p0.w, p1.x, p1.y, p1.z, p1.w};
      #pragma unroll
      for (int nf = 0; nf < 2; ++nf)
        #pragma unroll
        for (int j = 0; j < 8; ++j)
          acc[mf][nf][j] = fmaxf(acc[mf][nf][j], c[mf][nf][j] * inv[j]);
    }

    if (wave == 0) __builtin_amdgcn_s_wait_tensorcnt(0);   // next slice landed
    __syncthreads();                                       // all waves done with buffers
  }

  // scale by 1/nl, max-reduce over the 8 v's (8 adjacent lanes), store partial
  #pragma unroll
  for (int mf = 0; mf < 2; ++mf)
    #pragma unroll
    for (int nf = 0; nf < 2; ++nf)
      #pragma unroll
      for (int j = 0; j < 8; ++j) {
        float v = acc[mf][nf][j] * invnl[nf];
        v = fmaxf(v, __shfl_xor(v, 1, 32));
        v = fmaxf(v, __shfl_xor(v, 2, 32));
        v = fmaxf(v, __shfl_xor(v, 4, 32));
        if ((lane & 7) == 0) {
          int n = nbase + wave * 32 + nf * 16 + l15;
          int l = n >> 3;
          int bb = bbase + mf * 16 + hi * 8 + j;
          if (l < L_TOT)
            partial[((size_t)blockIdx.z * B_TOT + bb) * L_TOT + l] = v;
        }
      }
}

// ================= fallback: convert-in-kernel (round-1 style) =================
#define DPAD 776
__global__ __launch_bounds__(128, 1)
void sim_kernel_cvt(const float* __restrict__ embed, const float* __restrict__ labels,
                    const float* __restrict__ ne, const float* __restrict__ nl,
                    float* __restrict__ partial) {
  __shared__ alignas(16) bf16 sB[BN][DPAD];
  __shared__ alignas(16) bf16 sA[BM][DPAD];
  __shared__ alignas(16) float sInvNe[BM];

  const int tid   = threadIdx.x;
  const int lane  = tid & 31;
  const int wave  = tid >> 5;
  const int hi    = (lane >> 4) & 1;
  const int l15   = lane & 15;
  const int nbase = blockIdx.x * BN;
  const int bbase = blockIdx.y * BM;
  const int sbase = blockIdx.z * S_PER;

  for (int i = tid; i < BN * (D_DIM / 4); i += 128) {
    int r  = i / (D_DIM / 4);
    int c4 = i - r * (D_DIM / 4);
    int n  = nbase + r;
    float4 f = make_float4(0.f, 0.f, 0.f, 0.f);
    if (n < N_TOT) f = ((const float4*)(labels + (size_t)n * D_DIM))[c4];
    bf16x4 o; o[0] = (bf16)f.x; o[1] = (bf16)f.y; o[2] = (bf16)f.z; o[3] = (bf16)f.w;
    *(bf16x4*)&sB[r][c4 * 4] = o;
  }

  float invnl[2];
  #pragma unroll
  for (int nf = 0; nf < 2; ++nf) {
    int n = nbase + wave * 32 + nf * 16 + l15;
    invnl[nf] = (n < N_TOT) ? __builtin_amdgcn_rcpf(fmaxf(nl[n], EPS_N)) : 0.f;
  }

  float acc[2][2][8];
  #pragma unroll
  for (int mf = 0; mf < 2; ++mf)
    #pragma unroll
    for (int nf = 0; nf < 2; ++nf)
      #pragma unroll
      for (int j = 0; j < 8; ++j) acc[mf][nf][j] = -3.402823466e38f;

  for (int si = 0; si < S_PER; ++si) {
    const int s = sbase + si;
    __syncthreads();
    if (tid < BM)
      sInvNe[tid] = __builtin_amdgcn_rcpf(fmaxf(ne[(size_t)s * B_TOT + bbase + tid], EPS_N));
    for (int i = tid; i < BM * (D_DIM / 4); i += 128) {
      int r  = i / (D_DIM / 4);
      int c4 = i - r * (D_DIM / 4);
      float4 f = ((const float4*)(embed + ((size_t)s * B_TOT + bbase + r) * D_DIM))[c4];
      bf16x4 o; o[0] = (bf16)f.x; o[1] = (bf16)f.y; o[2] = (bf16)f.z; o[3] = (bf16)f.w;
      *(bf16x4*)&sA[r][c4 * 4] = o;
    }
    __syncthreads();

    v8f cz = {};
    v8f c[2][2]; c[0][0] = cz; c[0][1] = cz; c[1][0] = cz; c[1][1] = cz;

    #pragma unroll
    for (int kb = 0; kb < NKS; ++kb) {
      const int ka  = kb * 32 + hi * 8;
      const int kbb = kb * 32 + hi * 8;
      FragU a[2], b[2];
      #pragma unroll
      for (int mf = 0; mf < 2; ++mf) {
        const bf16* pr = &sA[mf * 16 + l15][0];
        a[mf].h[0] = *(const bf16x8*)(pr + ka);
        a[mf].h[1] = *(const bf16x8*)(pr + ka + 16);
      }
      #pragma unroll
      for (int nf = 0; nf < 2; ++nf) {
        const bf16* pr = &sB[wave * 32 + nf * 16 + l15][0];
        b[nf].h[0] = *(const bf16x8*)(pr + kbb);
        b[nf].h[1] = *(const bf16x8*)(pr + kbb + 16);
      }
      #pragma unroll
      for (int mf = 0; mf < 2; ++mf)
        #pragma unroll
        for (int nf = 0; nf < 2; ++nf)
          c[mf][nf] = __builtin_amdgcn_wmma_f32_16x16x32_bf16(
              false, a[mf].v, false, b[nf].v, (short)0, c[mf][nf], false, false);
    }

    #pragma unroll
    for (int mf = 0; mf < 2; ++mf) {
      const float4 p0 = *(const float4*)&sInvNe[mf * 16 + hi * 8];
      const float4 p1 = *(const float4*)&sInvNe[mf * 16 + hi * 8 + 4];
      float inv[8] = {p0.x, p0.y, p0.z, p0.w, p1.x, p1.y, p1.z, p1.w};
      #pragma unroll
      for (int nf = 0; nf < 2; ++nf)
        #pragma unroll
        for (int j = 0; j < 8; ++j)
          acc[mf][nf][j] = fmaxf(acc[mf][nf][j], c[mf][nf][j] * inv[j]);
    }
  }

  #pragma unroll
  for (int mf = 0; mf < 2; ++mf)
    #pragma unroll
    for (int nf = 0; nf < 2; ++nf)
      #pragma unroll
      for (int j = 0; j < 8; ++j) {
        float v = acc[mf][nf][j] * invnl[nf];
        v = fmaxf(v, __shfl_xor(v, 1, 32));
        v = fmaxf(v, __shfl_xor(v, 2, 32));
        v = fmaxf(v, __shfl_xor(v, 4, 32));
        if ((lane & 7) == 0) {
          int n = nbase + wave * 32 + nf * 16 + l15;
          int l = n >> 3;
          int bb = bbase + mf * 16 + hi * 8 + j;
          if (l < L_TOT)
            partial[((size_t)blockIdx.z * B_TOT + bb) * L_TOT + l] = v;
        }
      }
}

// ---------------- combine partial maxima over s-chunks, apply -1 floor ----------------
__global__ void combine_kernel(const float* __restrict__ partial, float* __restrict__ out) {
  int i = blockIdx.x * blockDim.x + threadIdx.x;
  if (i >= B_TOT * L_TOT) return;
  float v = -1.0f;
  #pragma unroll
  for (int sc = 0; sc < SCHUNKS; ++sc)
    v = fmaxf(v, partial[(size_t)sc * (B_TOT * L_TOT) + i]);
  out[i] = v;
}

extern "C" void kernel_launch(void* const* d_in, const int* in_sizes, int n_in,
                              void* d_out, int out_size, void* d_ws, size_t ws_size,
                              hipStream_t stream) {
  (void)in_sizes; (void)n_in; (void)out_size;
  const float* embed  = (const float*)d_in[0];   // [256,128,768] f32
  const float* labels = (const float*)d_in[1];   // [200,8,768]   f32

  float* ws  = (float*)d_ws;
  float* ne  = ws;                                   // 32768 floats
  float* nlp = ws + NE_CNT;                          // 1600 floats
  float* prt = ws + NE_CNT + N_TOT;                  // 204800 floats
  const size_t base_bytes = (size_t)(NE_CNT + N_TOT + PART_FLOATS) * 4;  // 956,672
  bf16* emb_bf = (bf16*)((char*)d_ws + base_bytes);
  bf16* lab_bf = emb_bf + EMB_ELEMS;
  const size_t need_fast = base_bytes + (EMB_ELEMS + LAB_ELEMS) * 2;     // ~53.7 MB

  norms_kernel<<<(NE_CNT + 7) / 8, 256, 0, stream>>>(embed, ne, NE_CNT);
  norms_kernel<<<(N_TOT + 7) / 8, 256, 0, stream>>>(labels, nlp, N_TOT);

  dim3 grid(NT, BT, SCHUNKS);   // 13 x 4 x 8 = 416 blocks

  if (ws_size >= need_fast) {
    int n8e = (int)(EMB_ELEMS / 8), n8l = (int)(LAB_ELEMS / 8);
    cvt_bf16_kernel<<<(n8e + 255) / 256, 256, 0, stream>>>(embed, emb_bf, n8e);
    cvt_bf16_kernel<<<(n8l + 255) / 256, 256, 0, stream>>>(labels, lab_bf, n8l);
    sim_kernel_tdm<<<grid, 128, 0, stream>>>(emb_bf, lab_bf, ne, nlp, prt);
  } else {
    sim_kernel_cvt<<<grid, 128, 0, stream>>>(embed, labels, ne, nlp, prt);
  }

  combine_kernel<<<(B_TOT * L_TOT + 255) / 256, 256, 0, stream>>>(prt, (float*)d_out);
}
